// MultiHeadAttentionBlock_21895743275358
// MI455X (gfx1250) — compile-verified, matrix operand order checked by reading, not run on hardware
//
#include <hip/hip_runtime.h>
#include <hip/hip_bf16.h>
#include <stdint.h>

// ---------------- problem constants ----------------
static constexpr int BB   = 4;      // batch
static constexpr int CC   = 2048;   // sequence length
static constexpr int DD   = 1024;   // model dim
static constexpr int HH   = 8;      // heads
static constexpr int KK   = 64;     // k_size
static constexpr int VV   = 128;    // v_size
static constexpr int NQ   = HH * KK;   // 512
static constexpr int NV   = HH * VV;   // 1024
static constexpr int MROWS = BB * CC;  // 8192
#define ATTN_SCALE 0.125f              // 1/sqrt(64)

typedef __attribute__((ext_vector_type(16))) _Float16 v16h;
typedef __attribute__((ext_vector_type(8)))  float    v8f;

#ifndef __has_builtin
#define __has_builtin(x) 0
#endif
#ifndef __has_include
#define __has_include(x) 0
#endif
#if defined(__AMDGCN__) && __has_builtin(__builtin_amdgcn_tensor_load_to_lds) && \
    __has_builtin(__builtin_amdgcn_s_wait_tensorcnt)
#define USE_TDM 1
// Toolchain arity probe: therock-10.0 headers (6-arg builtin) ship the TDM header.
#if __has_include(<hip/amd_detail/amd_gfx1250_TDM.h>)
#define TDM_6ARG 1
#else
#define TDM_6ARG 0
#endif
typedef __attribute__((ext_vector_type(4))) unsigned int u32x4;
typedef __attribute__((ext_vector_type(8))) int          i32x8;
typedef __attribute__((ext_vector_type(4))) int          i32x4;
#else
#define USE_TDM 0
#endif

// ---------------- WMMA helper (gfx1250, wave32) ----------------
__device__ __forceinline__ v8f wmma_f16(v16h a, v16h b, v8f c) {
    // v_wmma_f32_16x16x32_f16: D = A(16x32 f16) * B(32x16 f16) + C(16x16 f32)
    return __builtin_amdgcn_wmma_f32_16x16x32_f16(
        /*neg_a=*/false, a, /*neg_b=*/false, b,
        /*c_mod=*/(short)0, c, /*reuse_a=*/false, /*reuse_b=*/false);
}

// A-fragment loader: 16x32 f16 tile, row-major source, leading dim lda.
// ISA layout: lane L -> row m=L&15, half=L>>4; VGPR v holds K pair at
// kbase = (v>>2)*16 + half*8 + (v&3)*2.  The B layout is the transpose of the
// A layout, so a B-fragment of M^T is loaded with this same routine on M.
__device__ __forceinline__ v16h load_frag_row(const _Float16* __restrict__ base, int lda) {
    const int lane = threadIdx.x & 31;
    const int m = lane & 15, half = lane >> 4;
    const _Float16* p = base + (size_t)m * lda;
    v16h a;
#pragma unroll
    for (int v = 0; v < 8; ++v) {
        const int kb = ((v >> 2) << 4) + (half << 3) + ((v & 3) << 1);
        a[2 * v]     = p[kb];
        a[2 * v + 1] = p[kb + 1];
    }
    return a;
}

// B-fragment gather from a row-major KxN source (need T[n][k] = M[k][n]).
__device__ __forceinline__ v16h load_frag_col(const _Float16* __restrict__ base, int ld) {
    const int lane = threadIdx.x & 31;
    const int n = lane & 15, half = lane >> 4;
    v16h a;
#pragma unroll
    for (int v = 0; v < 8; ++v) {
        const int kb = ((v >> 2) << 4) + (half << 3) + ((v & 3) << 1);
        a[2 * v]     = base[(size_t)kb * ld + n];
        a[2 * v + 1] = base[(size_t)(kb + 1) * ld + n];
    }
    return a;
}

// ---------------- prep kernels ----------------
__global__ void cvt_f32_to_f16(const float* __restrict__ in, _Float16* __restrict__ out, int n) {
    int i = blockIdx.x * blockDim.x + threadIdx.x;
    if (i < n) out[i] = (_Float16)in[i];
}

// in: [D][N] f32 row-major  ->  out: [N][D] f16 row-major
__global__ void cvt_transpose(const float* __restrict__ in, _Float16* __restrict__ out,
                              int D, int N) {
    int i = blockIdx.x * blockDim.x + threadIdx.x;
    if (i < D * N) {
        int n = i / D, d = i % D;
        out[i] = (_Float16)in[(size_t)d * N + n];
    }
}

// ------- projection GEMM: C[M,N] = A[M,Kd] * Bt[N,Kd]^T, 16x64 tile / wave -------
__global__ __launch_bounds__(32)
void proj_gemm(const _Float16* __restrict__ A, const _Float16* __restrict__ Bt,
               _Float16* __restrict__ Cout, int M, int N, int Kd) {
    const int nt = blockIdx.x << 6;   // 64 output columns per wave
    const int mt = blockIdx.y << 4;   // 16 output rows
    const int lane = threadIdx.x & 31;
    v8f acc[4] = {};
    for (int k0 = 0; k0 < Kd; k0 += 32) {
        const v16h af = load_frag_row(A + (size_t)mt * Kd + k0, Kd);   // shared by 4 WMMAs
#pragma unroll
        for (int t = 0; t < 4; ++t) {
            v16h bf = load_frag_row(Bt + (size_t)(nt + 16 * t) * Kd + k0, Kd);
            acc[t] = wmma_f16(af, bf, acc[t]);
        }
    }
    const int n = lane & 15, half = lane >> 4;
#pragma unroll
    for (int t = 0; t < 4; ++t)
#pragma unroll
        for (int r = 0; r < 8; ++r)
            Cout[(size_t)(mt + r + 8 * half) * N + nt + 16 * t + n] = (_Float16)acc[t][r];
}

// ------- per-column softmax stats (softmax over the QUERY axis i, per reference) -------
__global__ __launch_bounds__(32)
void col_stats(const _Float16* __restrict__ Qh, const _Float16* __restrict__ Kh,
               float* __restrict__ cmax, float* __restrict__ csum) {
    const int bh = blockIdx.y;           // b*H + h
    const int jt = blockIdx.x << 4;      // column tile base
    const int lane = threadIdx.x & 31;
    const _Float16* Qb = Qh + (size_t)bh * CC * KK;   // [2048][64]
    const _Float16* Kb = Kh + (size_t)bh * CC * KK;

    // K fragments for this j-tile are loop-invariant
    const v16h kf0 = load_frag_row(Kb + (size_t)jt * KK + 0,  KK);
    const v16h kf1 = load_frag_row(Kb + (size_t)jt * KK + 32, KK);

    float m_run = -3.0e38f, s_run = 0.0f;
    for (int it = 0; it < CC; it += 16) {
        if (it + 16 < CC)
            __builtin_prefetch(Qb + (size_t)(it + 16) * KK, 0, 3);    // global_prefetch_b8
        v16h q0 = load_frag_row(Qb + (size_t)it * KK + 0,  KK);
        v16h q1 = load_frag_row(Qb + (size_t)it * KK + 32, KK);
        v8f acc = {};
        acc = wmma_f16(q0, kf0, acc);
        acc = wmma_f16(q1, kf1, acc);

        float sv[8], tm = -3.0e38f;
#pragma unroll
        for (int r = 0; r < 8; ++r) { sv[r] = acc[r] * ATTN_SCALE; tm = fmaxf(tm, sv[r]); }
        tm = fmaxf(tm, __shfl_xor(tm, 16, 32));   // rows 0-7 vs 8-15 share the column
        float ts = 0.0f;
#pragma unroll
        for (int r = 0; r < 8; ++r) ts += __expf(sv[r] - tm);
        ts += __shfl_xor(ts, 16, 32);
        const float mn = fmaxf(m_run, tm);        // online combine across i-tiles
        s_run = s_run * __expf(m_run - mn) + ts * __expf(tm - mn);
        m_run = mn;
    }
    if (lane < 16) {
        cmax[(size_t)bh * CC + jt + lane] = m_run;
        csum[(size_t)bh * CC + jt + lane] = s_run;
    }
}

// ------- attention output: out[i,:] = sum_j exp(S-cm[j])/cs[j] * V[j,:] -------
// One wave owns a full 16 x 128 output tile: S computed once per 32-wide j-step,
// V tiles streamed into LDS with the Tensor Data Mover (async), P staged via LDS.
__global__ __launch_bounds__(32)
void attn_out(const _Float16* __restrict__ Qh, const _Float16* __restrict__ Kh,
              const _Float16* __restrict__ Vh, const float* __restrict__ cmax,
              const float* __restrict__ csum, float* __restrict__ out) {
    __shared__ _Float16 P[16][32];        // probability tile (D-frag -> A-frag bounce)
    __shared__ _Float16 Vt[32][128];      // V tile staged by TDM (8 KB)
    const int it = blockIdx.x << 4;       // query tile
    const int bh = blockIdx.y;
    const int lane = threadIdx.x & 31;
    const int n = lane & 15, half = lane >> 4;

    const _Float16* Qb = Qh + (size_t)bh * CC * KK;
    const _Float16* Kb = Kh + (size_t)bh * CC * KK;
    const _Float16* Vb = Vh + (size_t)bh * CC * VV;
    const float* cm = cmax + (size_t)bh * CC;
    const float* cs = csum + (size_t)bh * CC;

    const v16h q0 = load_frag_row(Qb + (size_t)it * KK + 0,  KK);   // loop-invariant
    const v16h q1 = load_frag_row(Qb + (size_t)it * KK + 32, KK);

    v8f oacc[8] = {};                     // 16 x 128 fp32 accumulators
    for (int j0 = 0; j0 < CC; j0 += 32) {
#if USE_TDM
        {   // Async 32x128 f16 tile copy V[j0..j0+31][0..127] -> LDS via Tensor DMA.
            const uint64_t ga  = (uint64_t)(uintptr_t)(Vb + (size_t)j0 * VV);
            const uint32_t lds = (uint32_t)(uintptr_t)&Vt[0][0];
            u32x4 g0;
            g0[0] = 1u;                                   // count=1, user mode
            g0[1] = lds;                                  // lds_addr
            g0[2] = (uint32_t)ga;                         // global_addr[31:0]
            g0[3] = (uint32_t)((ga >> 32) & 0x01FFFFFFu)  // global_addr[56:32]
                    | 0x80000000u;                        // type=2 ("image")
            i32x8 g1;
            g1[0] = 1 << 16;                              // data_size=1 (2 bytes)
            g1[1] = (int)((unsigned)VV << 16);            // tensor_dim0[15:0]=128
            g1[2] = (int)((unsigned)CC << 16);            // dim0 hi=0 | tensor_dim1 lo=2048
            g1[3] = (int)((unsigned)VV << 16);            // dim1 hi=0 | tile_dim0=128
            g1[4] = 32;                                   // tile_dim1=32, tile_dim2=0
            g1[5] = VV;                                   // tensor_dim0_stride=128
            g1[6] = 0;                                    // stride0 hi | stride1 lo
            g1[7] = 0;                                    // stride1 hi
            const i32x4 gz4 = {0, 0, 0, 0};
#if TDM_6ARG
            const i32x8 gz8 = {0, 0, 0, 0, 0, 0, 0, 0};
            __builtin_amdgcn_tensor_load_to_lds(g0, g1, gz4, gz4, gz8, 0);
#else
            __builtin_amdgcn_tensor_load_to_lds(g0, g1, gz4, gz4, 0);
#endif
        }
#endif
        if (j0 + 32 < CC)
            __builtin_prefetch(Kb + (size_t)(j0 + 32) * KK, 0, 3);

        // S tile (16 x 32) -> probabilities -> LDS   (overlaps the TDM copy)
#pragma unroll
        for (int jj = 0; jj < 2; ++jj) {
            const int jb = j0 + (jj << 4);
            v16h k0 = load_frag_row(Kb + (size_t)jb * KK + 0,  KK);
            v16h k1 = load_frag_row(Kb + (size_t)jb * KK + 32, KK);
            v8f acc = {};
            acc = wmma_f16(q0, k0, acc);
            acc = wmma_f16(q1, k1, acc);
            const float cmv = cm[jb + n];
            const float rcs = 1.0f / cs[jb + n];
#pragma unroll
            for (int r = 0; r < 8; ++r) {
                const float p = __expf(acc[r] * ATTN_SCALE - cmv) * rcs;
                P[r + 8 * half][(jj << 4) + n] = (_Float16)p;
            }
        }
        __syncthreads();                   // order P ds_stores vs ds_loads
#if USE_TDM
        __builtin_amdgcn_s_wait_tensorcnt(0);
#endif
        const v16h pf = load_frag_row(&P[0][0], 32);      // P as A (16x32), one load
#pragma unroll
        for (int t = 0; t < 8; ++t) {
#if USE_TDM
            v16h vf = load_frag_col(&Vt[0][0] + 16 * t, VV);              // V^T from LDS
#else
            v16h vf = load_frag_col(Vb + (size_t)j0 * VV + 16 * t, VV);   // V^T from global
#endif
            oacc[t] = wmma_f16(pf, vf, oacc[t]);
        }
        __syncthreads();                   // drain ds reads before next TDM/P overwrite
    }
    // raw .view output order: flat = bh*C*V + i*V + v
    float* ob = out + (size_t)bh * CC * VV;
#pragma unroll
    for (int t = 0; t < 8; ++t)
#pragma unroll
        for (int r = 0; r < 8; ++r)
            ob[(size_t)(it + r + 8 * half) * VV + 16 * t + n] = oacc[t][r];
}

// ---------------- launcher ----------------
extern "C" void kernel_launch(void* const* d_in, const int* in_sizes, int n_in,
                              void* d_out, int out_size, void* d_ws, size_t ws_size,
                              hipStream_t stream) {
    const float* x  = (const float*)d_in[0];   // [4,2048,1024]
    const float* Mq = (const float*)d_in[1];   // [1024,512]
    const float* Mk = (const float*)d_in[2];   // [1024,512]
    const float* Mv = (const float*)d_in[3];   // [1024,1024]
    float* out = (float*)d_out;

    char* ws = (char*)d_ws;
    size_t off = 0;
    auto alloc = [&](size_t bytes) { char* p = ws + off; off += (bytes + 255) & ~size_t(255); return p; };
    _Float16* xh  = (_Float16*)alloc((size_t)MROWS * DD * 2);   // 16 MB
    _Float16* MqT = (_Float16*)alloc((size_t)NQ * DD * 2);      //  1 MB
    _Float16* MkT = (_Float16*)alloc((size_t)NQ * DD * 2);      //  1 MB
    _Float16* MvT = (_Float16*)alloc((size_t)NV * DD * 2);      //  2 MB
    _Float16* Qh  = (_Float16*)alloc((size_t)MROWS * NQ * 2);   //  8 MB
    _Float16* Kh  = (_Float16*)alloc((size_t)MROWS * NQ * 2);   //  8 MB
    _Float16* Vh  = (_Float16*)alloc((size_t)MROWS * NV * 2);   // 16 MB
    float*    cmx = (float*)alloc((size_t)BB * HH * CC * 4);    // 256 KB
    float*    csm = (float*)alloc((size_t)BB * HH * CC * 4);    // 256 KB

    // --- precision prep: fp32 -> f16 (weights transposed to [N][D]) ---
    {
        int n = MROWS * DD;
        cvt_f32_to_f16<<<(n + 255) / 256, 256, 0, stream>>>(x, xh, n);
        n = DD * NQ;
        cvt_transpose<<<(n + 255) / 256, 256, 0, stream>>>(Mq, MqT, DD, NQ);
        cvt_transpose<<<(n + 255) / 256, 256, 0, stream>>>(Mk, MkT, DD, NQ);
        n = DD * NV;
        cvt_transpose<<<(n + 255) / 256, 256, 0, stream>>>(Mv, MvT, DD, NV);
    }

    // --- projections (register-blocked 16x64 tiles, WMMA f16 -> fp32 acc) ---
    {
        dim3 blk(32);
        proj_gemm<<<dim3(NQ / 64, MROWS / 16), blk, 0, stream>>>(xh, MqT, Qh, MROWS, NQ, DD);
        proj_gemm<<<dim3(NQ / 64, MROWS / 16), blk, 0, stream>>>(xh, MkT, Kh, MROWS, NQ, DD);
        proj_gemm<<<dim3(NV / 64, MROWS / 16), blk, 0, stream>>>(xh, MvT, Vh, MROWS, NV, DD);
    }

    // --- column softmax stats (softmax over query axis, per the reference) ---
    col_stats<<<dim3(CC / 16, BB * HH), dim3(32), 0, stream>>>(Qh, Kh, cmx, csm);

    // --- second pass: S recompute + P*V, one wave per 16x128 output tile ---
    attn_out<<<dim3(CC / 16, BB * HH), dim3(32), 0, stream>>>(Qh, Kh, Vh, cmx, csm, out);

    (void)in_sizes; (void)n_in; (void)out_size; (void)ws_size;
}
